// MixerBlock_19756849562055
// MI455X (gfx1250) — matrix operand order repaired and checked by reading, not compiled, not run on hardware
//
#include <hip/hip_runtime.h>
#include <cstdint>
#include <cstddef>

// ---------------------------------------------------------------------------
// MixerBlock for MI455X (gfx1250, wave32, WMMA).
// All GEMMs: bf16 A/B with fp32 accumulate via v_wmma_f32_16x16x32_bf16.
// K-loop is software-pipelined (double-buffered fragments) with unconditional
// speculative global_prefetch.
// ---------------------------------------------------------------------------

#define B_   4
#define N_   12288
#define C_   256
#define KCH_ 6
#define NB_  8
#define T_   512
#define CH_  1024
#define KC_  (KCH_ * C_)   // 1536
#define BN_  (B_ * N_)     // 49152

typedef __attribute__((ext_vector_type(16))) __bf16 v16bf;
typedef __attribute__((ext_vector_type(8)))  float  v8f;

union Frag { uint4 u4[2]; v16bf v; };

__device__ __forceinline__ unsigned short f2bf(float f) {
  unsigned int x = __builtin_bit_cast(unsigned int, f);
  x += 0x7FFFu + ((x >> 16) & 1u);            // round-to-nearest-even
  return (unsigned short)(x >> 16);
}

__device__ __forceinline__ float gelu_tanh(float x) {
  float x3 = x * x * x;
  float t  = tanhf(0.7978845608028654f * (x + 0.044715f * x3));
  return 0.5f * x * (1.0f + t);
}

// ---------------------------------------------------------------------------
// Pack fp32 row-major W[Kd,Nd] into bf16 B-fragment images.
// Tile (kt,nt): K rows kt*32..+31, N cols nt*16..+15.
// Lane l (l<16): K = 2j,2j+1 at col l. Lane l>=16: K = 16+2j,17+2j at col l-16.
// Stored [tile][lane][8 dwords] -> GEMM does two contiguous b128 loads/lane.
// ---------------------------------------------------------------------------
__global__ __launch_bounds__(256)
void pack_b_kernel(const float* __restrict__ W, unsigned int* __restrict__ P,
                   int Kd, int Nd) {
  int tid  = blockIdx.x * 256 + threadIdx.x;
  int lane = tid & 31;
  int tile = tid >> 5;
  int NT   = Nd >> 4;
  int kt   = tile / NT;
  int nt   = tile - kt * NT;
  int k0   = kt * 32 + ((lane >> 4) << 4);
  int n    = nt * 16 + (lane & 15);
  size_t base = (size_t)tid * 8;
#pragma unroll
  for (int j = 0; j < 8; ++j) {
    float lo = W[(size_t)(k0 + 2 * j)     * Nd + n];
    float hi = W[(size_t)(k0 + 2 * j + 1) * Nd + n];
    P[base + j] = (unsigned int)f2bf(lo) | ((unsigned int)f2bf(hi) << 16);
  }
}

// ---------------------------------------------------------------------------
// LayerNorm over C=256. One wave per row (8 floats per lane).
// ---------------------------------------------------------------------------
__global__ __launch_bounds__(256)
void ln_kernel(const float* __restrict__ X, const float* __restrict__ sc,
               const float* __restrict__ bs, float* __restrict__ outf,
               unsigned short* __restrict__ outb, int ldb, int coloff,
               int transposed) {
  int lane = threadIdx.x & 31;
  int row  = blockIdx.x * 8 + (threadIdx.x >> 5);
  int c0   = lane * 8;
  const float* xr = X + (size_t)row * C_ + c0;
  float4 a0 = *(const float4*)xr;
  float4 a1 = *(const float4*)(xr + 4);
  float v[8] = {a0.x, a0.y, a0.z, a0.w, a1.x, a1.y, a1.z, a1.w};
  float s = 0.f, q = 0.f;
#pragma unroll
  for (int i = 0; i < 8; ++i) { s += v[i]; q += v[i] * v[i]; }
#pragma unroll
  for (int off = 16; off >= 1; off >>= 1) {
    s += __shfl_xor(s, off, 32);
    q += __shfl_xor(q, off, 32);
  }
  float mean = s * (1.0f / C_);
  float var  = q * (1.0f / C_) - mean * mean;
  float rstd = rsqrtf(var + 1e-6f);
#pragma unroll
  for (int i = 0; i < 8; ++i)
    v[i] = (v[i] - mean) * rstd * sc[c0 + i] + bs[c0 + i];
  if (outf) {
    float* o = outf + (size_t)row * C_ + c0;
    *(float4*)o       = float4{v[0], v[1], v[2], v[3]};
    *(float4*)(o + 4) = float4{v[4], v[5], v[6], v[7]};
  }
  if (outb) {
    if (!transposed) {
      unsigned int u0 = (unsigned int)f2bf(v[0]) | ((unsigned int)f2bf(v[1]) << 16);
      unsigned int u1 = (unsigned int)f2bf(v[2]) | ((unsigned int)f2bf(v[3]) << 16);
      unsigned int u2 = (unsigned int)f2bf(v[4]) | ((unsigned int)f2bf(v[5]) << 16);
      unsigned int u3 = (unsigned int)f2bf(v[6]) | ((unsigned int)f2bf(v[7]) << 16);
      *(uint4*)(outb + (size_t)row * ldb + coloff + c0) = uint4{u0, u1, u2, u3};
    } else {
      int b = row / N_;
      int n = row - b * N_;
#pragma unroll
      for (int i = 0; i < 8; ++i)
        outb[(size_t)(b * C_ + c0 + i) * N_ + n] = f2bf(v[i]);
    }
  }
}

// ---------------------------------------------------------------------------
// Chebyshev graph-Laplacian apply (gathers are L2-resident: 50MB << 192MB L2).
//   tnext = alpha*(sum_k nw[n,k]*tcur[b,idx[n,k],:] + dw[n]*tcur[b,n,:])
//           + beta*tprev[b,n,:];   also bf16 slice -> xk slot.
// ---------------------------------------------------------------------------
__global__ __launch_bounds__(256)
void lapply_kernel(const float* __restrict__ tcur, const float* __restrict__ tprev,
                   const int* __restrict__ nidx, const float* __restrict__ nw,
                   const float* __restrict__ dw, float alpha, float beta,
                   float* __restrict__ tnext, unsigned short* __restrict__ xk,
                   int slot) {
  int lane = threadIdx.x & 31;
  int row  = blockIdx.x * 8 + (threadIdx.x >> 5);
  int b = row / N_;
  int n = row - b * N_;
  int c0 = lane * 8;
  const float* batch = tcur + (size_t)b * N_ * C_ + c0;
  float acc[8];
  {
    const float* p = tcur + (size_t)row * C_ + c0;
    float4 g0 = *(const float4*)p;
    float4 g1 = *(const float4*)(p + 4);
    float d = dw[n];
    acc[0] = d * g0.x; acc[1] = d * g0.y; acc[2] = d * g0.z; acc[3] = d * g0.w;
    acc[4] = d * g1.x; acc[5] = d * g1.y; acc[6] = d * g1.z; acc[7] = d * g1.w;
  }
#pragma unroll
  for (int k = 0; k < NB_; ++k) {
    int   j  = nidx[n * NB_ + k];
    float wv = nw[n * NB_ + k];
    const float* p = batch + (size_t)j * C_;
    float4 g0 = *(const float4*)p;
    float4 g1 = *(const float4*)(p + 4);
    acc[0] += wv * g0.x; acc[1] += wv * g0.y; acc[2] += wv * g0.z; acc[3] += wv * g0.w;
    acc[4] += wv * g1.x; acc[5] += wv * g1.y; acc[6] += wv * g1.z; acc[7] += wv * g1.w;
  }
  const float* pp = tprev + (size_t)row * C_ + c0;
  float4 p0 = *(const float4*)pp;
  float4 p1 = *(const float4*)(pp + 4);
  float pv[8] = {p0.x, p0.y, p0.z, p0.w, p1.x, p1.y, p1.z, p1.w};
  float r[8];
#pragma unroll
  for (int i = 0; i < 8; ++i) r[i] = alpha * acc[i] + beta * pv[i];
  float* o = tnext + (size_t)row * C_ + c0;
  *(float4*)o       = float4{r[0], r[1], r[2], r[3]};
  *(float4*)(o + 4) = float4{r[4], r[5], r[6], r[7]};
  unsigned int u0 = (unsigned int)f2bf(r[0]) | ((unsigned int)f2bf(r[1]) << 16);
  unsigned int u1 = (unsigned int)f2bf(r[2]) | ((unsigned int)f2bf(r[3]) << 16);
  unsigned int u2 = (unsigned int)f2bf(r[4]) | ((unsigned int)f2bf(r[5]) << 16);
  unsigned int u3 = (unsigned int)f2bf(r[6]) | ((unsigned int)f2bf(r[7]) << 16);
  *(uint4*)(xk + (size_t)row * KC_ + slot * C_ + c0) = uint4{u0, u1, u2, u3};
}

// ---------------------------------------------------------------------------
// WMMA GEMM, software-pipelined. Workgroup = 8 waves (2M x 4N);
// wave tile = 32x64 = 2x4 WMMA tiles; nk (=Kd/32) is even and >= 8 here.
// mode 0: outf = gelu(acc+bias)        mode 1: outb = bf16(gelu(acc+bias))
// mode 2: transposed residual add      mode 3: row-major residual add
// ---------------------------------------------------------------------------
__global__ __launch_bounds__(256)
void gemm_bf16_wmma(const unsigned short* __restrict__ A,
                    const unsigned int* __restrict__ Bp,
                    const float* __restrict__ bias,
                    float* __restrict__ outf, unsigned short* __restrict__ outb,
                    const float* __restrict__ res, float* __restrict__ resout,
                    int M, int Nd, int Kd, int ldA, int mode) {
  int lane  = threadIdx.x & 31;
  int wave  = threadIdx.x >> 5;
  int wm    = wave >> 2;           // 0..1
  int wn    = wave & 3;            // 0..3
  int mwave = blockIdx.y * 64 + wm * 32;
  int nwave = blockIdx.x * 256 + wn * 64;
  int lhalf = lane >> 4;           // 0|1
  int lmod  = lane & 15;
  int NT    = Nd >> 4;
  int nk    = Kd >> 5;
  int bStep = NT * 64;             // uint4 stride per k-tile in packed B

  // Per-lane fragment base pointers (uint4 units).
  const uint4* aP[2];
  const uint4* bP[4];
#pragma unroll
  for (int mi = 0; mi < 2; ++mi)
    aP[mi] = (const uint4*)(A + (size_t)(mwave + mi * 16 + lmod) * ldA + lhalf * 16);
#pragma unroll
  for (int ni = 0; ni < 4; ++ni)
    bP[ni] = (const uint4*)Bp + (size_t)((nwave + ni * 16) >> 4) * 64 + lane * 2;

  v8f acc[2][4];
#pragma unroll
  for (int mi = 0; mi < 2; ++mi)
#pragma unroll
    for (int ni = 0; ni < 4; ++ni) acc[mi][ni] = (v8f)(0.0f);

  auto loadFrags = [&](Frag* a, Frag* b, int kt) {
#pragma unroll
    for (int mi = 0; mi < 2; ++mi) {
      const uint4* p = aP[mi] + (size_t)kt * 4;   // 64B per k-tile
      a[mi].u4[0] = p[0];
      a[mi].u4[1] = p[1];
    }
#pragma unroll
    for (int ni = 0; ni < 4; ++ni) {
      const uint4* q = bP[ni] + (size_t)kt * bStep;
      b[ni].u4[0] = q[0];
      b[ni].u4[1] = q[1];
    }
  };
  auto wmma8 = [&](Frag* a, Frag* b) {
#pragma unroll
    for (int mi = 0; mi < 2; ++mi)
#pragma unroll
      for (int ni = 0; ni < 4; ++ni)
        acc[mi][ni] = __builtin_amdgcn_wmma_f32_16x16x32_bf16(
            false, a[mi].v, false, b[ni].v, (short)0, acc[mi][ni], false, false);
  };

  Frag a0[2], b0[4], a1[2], b1[4];
  loadFrags(a0, b0, 0);
  int kt = 0;
  for (; kt < nk - 2; kt += 2) {
    loadFrags(a1, b1, kt + 1);
    // Unconditional speculative prefetch 4 k-tiles ahead (256B) -> no branches.
    __builtin_prefetch(aP[0] + (size_t)(kt + 4) * 4, 0, 1);
    __builtin_prefetch(aP[1] + (size_t)(kt + 4) * 4, 0, 1);
    wmma8(a0, b0);
    loadFrags(a0, b0, kt + 2);
    wmma8(a1, b1);
  }
  loadFrags(a1, b1, nk - 1);  // kt == nk-2 here; both tail tiles in-bounds
  wmma8(a0, b0);
  wmma8(a1, b1);

  // Epilogue. C layout: VGPR r, lanes 0-15 -> M=r, lanes 16-31 -> M=8+r; N=lane&15
#pragma unroll
  for (int mi = 0; mi < 2; ++mi) {
#pragma unroll
    for (int ni = 0; ni < 4; ++ni) {
      int m0 = mwave + mi * 16;
      int n  = nwave + ni * 16 + lmod;
      float bv = bias[n];
#pragma unroll
      for (int r = 0; r < 8; ++r) {
        int m = m0 + r + lhalf * 8;
        float v = acc[mi][ni][r] + bv;
        if (mode == 0) {
          outf[(size_t)m * Nd + n] = gelu_tanh(v);
        } else if (mode == 1) {
          outb[(size_t)m * Nd + n] = f2bf(gelu_tanh(v));
        } else if (mode == 2) {
          int bb = m >> 8;             // m / C_
          int cc = m & 255;            // m % C_
          size_t idx = ((size_t)bb * N_ + n) * C_ + cc;
          resout[idx] = res[idx] + v;
        } else {
          size_t idx = (size_t)m * Nd + n;
          resout[idx] = res[idx] + v;
        }
      }
    }
  }
}

// ---------------------------------------------------------------------------
extern "C" void kernel_launch(void* const* d_in, const int* in_sizes, int n_in,
                              void* d_out, int out_size, void* d_ws, size_t ws_size,
                              hipStream_t stream) {
  (void)in_sizes; (void)n_in; (void)out_size; (void)ws_size;
  const float* maps   = (const float*)d_in[0];
  const int*   nidx   = (const int*)d_in[1];
  const float* nw     = (const float*)d_in[2];
  const float* dw     = (const float*)d_in[3];
  const float* ln1s   = (const float*)d_in[4];
  const float* ln1b   = (const float*)d_in[5];
  const float* conv1w = (const float*)d_in[6];
  const float* conv1b = (const float*)d_in[7];
  const float* ln2s   = (const float*)d_in[8];
  const float* ln2b   = (const float*)d_in[9];
  const float* tokw1  = (const float*)d_in[10];
  const float* tokb1  = (const float*)d_in[11];
  const float* tokw2  = (const float*)d_in[12];
  const float* tokb2  = (const float*)d_in[13];
  const float* ln3s   = (const float*)d_in[14];
  const float* ln3b   = (const float*)d_in[15];
  const float* conv2w = (const float*)d_in[16];
  const float* conv2b = (const float*)d_in[17];
  const float* ln4s   = (const float*)d_in[18];
  const float* ln4b   = (const float*)d_in[19];
  const float* chw1   = (const float*)d_in[20];
  const float* chb1   = (const float*)d_in[21];
  const float* chw2   = (const float*)d_in[22];
  const float* chb2   = (const float*)d_in[23];

  uint8_t* w = (uint8_t*)d_ws;
  const size_t FB = (size_t)BN_ * C_ * sizeof(float);        // 50,331,648 B
  float* bufA = (float*)(w);
  float* bufB = (float*)(w + FB);
  float* bufC = (float*)(w + 2 * FB);
  float* bufD = (float*)(w + 3 * FB);
  unsigned short* xk = (unsigned short*)(w + 4 * FB);        // BN x KC bf16
  const size_t XKB = (size_t)BN_ * KC_ * 2;                  // 150,994,944 B
  uint8_t* pw = w + 4 * FB + XKB;
  const size_t CVB = (size_t)KC_ * C_ * 2;
  const size_t T1B = (size_t)N_ * T_ * 2;
  const size_t C1B = (size_t)C_ * CH_ * 2;
  unsigned int* conv1p = (unsigned int*)(pw);
  unsigned int* conv2p = (unsigned int*)(pw + CVB);
  unsigned int* tok1p  = (unsigned int*)(pw + 2 * CVB);
  unsigned int* tok2p  = (unsigned int*)(pw + 2 * CVB + T1B);
  unsigned int* ch1p   = (unsigned int*)(pw + 2 * CVB + 2 * T1B);
  unsigned int* ch2p   = (unsigned int*)(pw + 2 * CVB + 2 * T1B + C1B);

  unsigned short* atok = (unsigned short*)bufA;  // (B*C, N) bf16 token-mix A
  unsigned short* htok = (unsigned short*)bufB;  // (B*C, T) bf16 token hidden
  unsigned short* lnch = (unsigned short*)bufB;  // (BN, C) bf16 channel A
  unsigned short* hch  = (unsigned short*)xk;    // (BN, CH) bf16 channel hidden

  auto pack = [&](const float* W, unsigned int* P, int Kd, int Nd) {
    int threads = (Kd / 32) * (Nd / 16) * 32;
    pack_b_kernel<<<threads / 256, 256, 0, stream>>>(W, P, Kd, Nd);
  };
  auto gemm = [&](const unsigned short* A, const unsigned int* Bp,
                  const float* bias, float* outf, unsigned short* outb,
                  const float* res, float* resout, int M, int Nd, int Kd,
                  int ldA, int mode) {
    dim3 grid(Nd / 256, M / 64);
    gemm_bf16_wmma<<<grid, 256, 0, stream>>>(A, Bp, bias, outf, outb, res,
                                             resout, M, Nd, Kd, ldA, mode);
  };
  const int LNG = BN_ / 8;   // 6144 blocks of 256

  // Weight packing (once per launch; deterministic)
  pack(conv1w, conv1p, KC_, C_);
  pack(conv2w, conv2p, KC_, C_);
  pack(tokw1,  tok1p,  N_,  T_);
  pack(tokw2,  tok2p,  T_,  N_);
  pack(chw1,   ch1p,   C_,  CH_);
  pack(chw2,   ch2p,   CH_, C_);

  // --- stage 1: y = LN1(x); also xk slot 0 = bf16(y) ---
  ln_kernel<<<LNG, 256, 0, stream>>>(maps, ln1s, ln1b, bufA, xk, KC_, 0, 0);
  // Chebyshev recurrence -> xk slots 1..5
  lapply_kernel<<<LNG, 256, 0, stream>>>(bufA, bufA, nidx, nw, dw, 1.f,  0.f, bufB, xk, 1);
  lapply_kernel<<<LNG, 256, 0, stream>>>(bufB, bufA, nidx, nw, dw, 2.f, -1.f, bufC, xk, 2);
  lapply_kernel<<<LNG, 256, 0, stream>>>(bufC, bufB, nidx, nw, dw, 2.f, -1.f, bufA, xk, 3);
  lapply_kernel<<<LNG, 256, 0, stream>>>(bufA, bufC, nidx, nw, dw, 2.f, -1.f, bufB, xk, 4);
  lapply_kernel<<<LNG, 256, 0, stream>>>(bufB, bufA, nidx, nw, dw, 2.f, -1.f, bufC, xk, 5);
  // conv1: gelu(xk @ W1 + b1) -> bufD (fp32)
  gemm(xk, conv1p, conv1b, bufD, nullptr, nullptr, nullptr, BN_, C_, KC_, KC_, 0);
  // LN2 -> transposed bf16 A (B*C, N)
  ln_kernel<<<LNG, 256, 0, stream>>>(bufD, ln2s, ln2b, nullptr, atok, 0, 0, 1);
  // token MLP
  gemm(atok, tok1p, tokb1, nullptr, htok, nullptr, nullptr, B_ * C_, T_, N_, N_, 1);
  gemm(htok, tok2p, tokb2, nullptr, nullptr, maps, bufD, B_ * C_, N_, T_, T_, 2); // bufD = x_res
  // --- stage 2 ---
  ln_kernel<<<LNG, 256, 0, stream>>>(bufD, ln3s, ln3b, bufA, xk, KC_, 0, 0);
  lapply_kernel<<<LNG, 256, 0, stream>>>(bufA, bufA, nidx, nw, dw, 1.f,  0.f, bufB, xk, 1);
  lapply_kernel<<<LNG, 256, 0, stream>>>(bufB, bufA, nidx, nw, dw, 2.f, -1.f, bufC, xk, 2);
  lapply_kernel<<<LNG, 256, 0, stream>>>(bufC, bufB, nidx, nw, dw, 2.f, -1.f, bufA, xk, 3);
  lapply_kernel<<<LNG, 256, 0, stream>>>(bufA, bufC, nidx, nw, dw, 2.f, -1.f, bufB, xk, 4);
  lapply_kernel<<<LNG, 256, 0, stream>>>(bufB, bufA, nidx, nw, dw, 2.f, -1.f, bufC, xk, 5);
  gemm(xk, conv2p, conv2b, bufA, nullptr, nullptr, nullptr, BN_, C_, KC_, KC_, 0);
  ln_kernel<<<LNG, 256, 0, stream>>>(bufA, ln4s, ln4b, nullptr, lnch, C_, 0, 0);
  // channel MLP + final residual
  gemm(lnch, ch1p, chb1, nullptr, hch, nullptr, nullptr, BN_, CH_, C_, C_, 1);
  gemm(hch,  ch2p, chb2, nullptr, nullptr, bufD, (float*)d_out, BN_, C_, CH_, CH_, 3);
}
